// NTXentLoss_50749333569628
// MI455X (gfx1250) — matrix operand order repaired and checked by reading, not compiled, not run on hardware
//
#include <hip/hip_runtime.h>
#include <hip/hip_bf16.h>

typedef __attribute__((ext_vector_type(16))) _Float16 v16h;
typedef __attribute__((ext_vector_type(8)))  float    v8f;
typedef __attribute__((ext_vector_type(4)))  unsigned int u32x4;
typedef __attribute__((ext_vector_type(4)))  int          i32x4;

#define D_DIM 512
#define TEMP_INV 2.0f        // 1 / 0.5
#define TILE 128             // workgroup output tile: 128x128
#define KCHUNK 32            // K per WMMA
#define NCHUNK (D_DIM / KCHUNK)
#define LDA_H 40             // LDS row stride in halves (32 + 8 pad)

#if __has_builtin(__builtin_amdgcn_global_load_async_to_lds_b128)
#define HAVE_ASYNC_LDS 1
typedef __attribute__((address_space(1))) i32x4 gi32x4;   // global int4
typedef __attribute__((address_space(3))) i32x4 li32x4;   // LDS int4
#if __has_builtin(__builtin_amdgcn_s_wait_asynccnt)
#define WAIT_ASYNC(n) __builtin_amdgcn_s_wait_asynccnt(n)
#else
#define WAIT_ASYNC(n) asm volatile("s_wait_asynccnt " #n ::: "memory")
#endif
#endif

// ---------------------------------------------------------------------------
// Kernel 1: per-pair norms, cosine positive pairs, normalized f16 matrix,
// and zero-init of neg[] accumulators.
// ---------------------------------------------------------------------------
__global__ __launch_bounds__(256) void ntxent_prep(
    const float* __restrict__ y, const float* __restrict__ yh,
    _Float16* __restrict__ Xn, float* __restrict__ pos,
    float* __restrict__ neg, int N)
{
    const int i = blockIdx.x;
    const int t = threadIdx.x;
    const float* yr  = y  + (size_t)i * D_DIM;
    const float* yhr = yh + (size_t)i * D_DIM;

    float a0 = yr[t],  a1 = yr[t + 256];
    float b0 = yhr[t], b1 = yhr[t + 256];

    __shared__ float sA[256], sB[256], sC[256];
    sA[t] = a0 * a0 + a1 * a1;
    sB[t] = b0 * b0 + b1 * b1;
    sC[t] = a0 * b0 + a1 * b1;
    __syncthreads();
    for (int off = 128; off > 0; off >>= 1) {
        if (t < off) {
            sA[t] += sA[t + off];
            sB[t] += sB[t + off];
            sC[t] += sC[t + off];
        }
        __syncthreads();
    }

    __shared__ float s_invy, s_invyh;
    if (t == 0) {
        float ny  = sqrtf(sA[0]);
        float nyh = sqrtf(sB[0]);
        s_invy  = 1.0f / fmaxf(ny,  1e-12f);   // F.normalize eps
        s_invyh = 1.0f / fmaxf(nyh, 1e-12f);
        float cosv = sC[0] / (fmaxf(ny, 1e-8f) * fmaxf(nyh, 1e-8f)); // cosine eps
        pos[i] = __expf(cosv * TEMP_INV);
        neg[i] = 0.0f;
        neg[i + N] = 0.0f;
    }
    __syncthreads();

    const float iy = s_invy, iyh = s_invyh;
    _Float16* r0 = Xn + (size_t)i * D_DIM;
    _Float16* r1 = Xn + (size_t)(i + N) * D_DIM;
    r0[t]        = (_Float16)(a0 * iy);
    r0[t + 256]  = (_Float16)(a1 * iy);
    r1[t]        = (_Float16)(b0 * iyh);
    r1[t + 256]  = (_Float16)(b1 * iyh);
}

// ---------------------------------------------------------------------------
// Kernel 2: fused  exp(Xn Xn^T / T)  row-sum with diagonal removed.
// ---------------------------------------------------------------------------
union FragH { v16h h; u32x4 q[2]; };

// one K-chunk of WMMAs from staged LDS panels
__device__ __forceinline__ void mma_chunk(
    const _Float16* __restrict__ pa, const _Float16* __restrict__ pb,
    v8f c[4][2], int wr, int wc, int lm, int lhi)
{
    // A fragments per ISA 16-bit A 16x32 table: lane-half picks K-base {0,8};
    // halves 0..7 -> K=kb..kb+7, halves 8..15 -> K=16+kb..16+kb+7.
    FragH a[4];
    const int kbA = lhi * 8;
#pragma unroll
    for (int m = 0; m < 4; ++m) {
        const _Float16* base = pa + (wr * 64 + m * 16 + lm) * LDA_H;
        a[m].q[0] = *(const u32x4*)(base + kbA);
        a[m].q[1] = *(const u32x4*)(base + 16 + kbA);
    }
    // B fragments: same K order as A: lane-half picks K-base {0,8};
    // halves 0..7 -> K=kb..kb+7, halves 8..15 -> K=16+kb..16+kb+7. N = lane%16. B = Xn^T -> read rows.
    FragH b[2];
    const int kbB = lhi * 8;
#pragma unroll
    for (int n = 0; n < 2; ++n) {
        const _Float16* base = pb + (wc * 32 + n * 16 + lm) * LDA_H;
        b[n].q[0] = *(const u32x4*)(base + kbB);
        b[n].q[1] = *(const u32x4*)(base + 16 + kbB);
    }
#pragma unroll
    for (int m = 0; m < 4; ++m)
#pragma unroll
        for (int n = 0; n < 2; ++n)
            c[m][n] = __builtin_amdgcn_wmma_f32_16x16x32_f16(
                false, a[m].h, false, b[n].h,
                (short)0, c[m][n], false, false);
}

__global__ __launch_bounds__(256) void ntxent_simrows(
    const _Float16* __restrict__ Xn, float* __restrict__ neg, int Tt)
{
#ifdef HAVE_ASYNC_LDS
    __shared__ _Float16 ldsA[2][TILE * LDA_H];
    __shared__ _Float16 ldsB[2][TILE * LDA_H];
#else
    __shared__ _Float16 ldsA[1][TILE * LDA_H];
    __shared__ _Float16 ldsB[1][TILE * LDA_H];
#endif

    // decode upper-triangular tile pair (rt <= ct)
    int rem = blockIdx.x;
    int rt = 0;
    while (rem >= Tt - rt) { rem -= (Tt - rt); ++rt; }
    const int ct = rt + rem;

    const int tid  = threadIdx.x;
    const int wid  = tid >> 5;
    const int lane = tid & 31;
    const int wr   = wid >> 2;          // 0..1 : 64-row strip
    const int wc   = wid & 3;           // 0..3 : 32-col strip
    const int lm   = lane & 15;
    const int lhi  = lane >> 4;         // 0 or 1 (lane half)

    const int rowA = rt * TILE;
    const int rowB = ct * TILE;

    // this thread's staging slice: 2 rows x 8 halves per matrix per chunk
    const int lid0 = tid, lid1 = tid + 256;
    const int srow0 = lid0 >> 2, scol0 = (lid0 & 3) << 3;
    const int srow1 = lid1 >> 2, scol1 = (lid1 & 3) << 3;

    v8f c[4][2];
#pragma unroll
    for (int m = 0; m < 4; ++m)
#pragma unroll
        for (int n = 0; n < 2; ++n)
            c[m][n] = (v8f){};

#ifdef HAVE_ASYNC_LDS
    // ---- double-buffered async global->LDS pipeline (ASYNCcnt) ----
    #define STAGE_ASYNC(K0, BUF)                                                \
        do {                                                                    \
            __builtin_amdgcn_global_load_async_to_lds_b128(                     \
                (gi32x4*)(Xn + (size_t)(rowA + srow0) * D_DIM + (K0) + scol0),  \
                (li32x4*)&ldsA[BUF][srow0 * LDA_H + scol0], 0, 0);              \
            __builtin_amdgcn_global_load_async_to_lds_b128(                     \
                (gi32x4*)(Xn + (size_t)(rowB + srow0) * D_DIM + (K0) + scol0),  \
                (li32x4*)&ldsB[BUF][srow0 * LDA_H + scol0], 0, 0);              \
            __builtin_amdgcn_global_load_async_to_lds_b128(                     \
                (gi32x4*)(Xn + (size_t)(rowA + srow1) * D_DIM + (K0) + scol1),  \
                (li32x4*)&ldsA[BUF][srow1 * LDA_H + scol1], 0, 0);              \
            __builtin_amdgcn_global_load_async_to_lds_b128(                     \
                (gi32x4*)(Xn + (size_t)(rowB + srow1) * D_DIM + (K0) + scol1),  \
                (li32x4*)&ldsB[BUF][srow1 * LDA_H + scol1], 0, 0);              \
        } while (0)

    STAGE_ASYNC(0, 0);
    for (int k = 0; k < NCHUNK; ++k) {
        const int cur = k & 1;
        if (k < NCHUNK - 1) {
            STAGE_ASYNC((k + 1) * KCHUNK, cur ^ 1);
            WAIT_ASYNC(4);   // in-order retire: current buffer's 4 loads done
        } else {
            WAIT_ASYNC(0);
        }
        __syncthreads();     // other waves' async writes also landed
        mma_chunk(ldsA[cur], ldsB[cur], c, wr, wc, lm, lhi);
        __syncthreads();     // all reads done before buffer is overwritten
    }
#else
    // ---- fallback: synchronous staged copy ----
    for (int k0 = 0; k0 < D_DIM; k0 += KCHUNK) {
        u32x4 a0 = *(const u32x4*)(Xn + (size_t)(rowA + srow0) * D_DIM + k0 + scol0);
        u32x4 b0 = *(const u32x4*)(Xn + (size_t)(rowB + srow0) * D_DIM + k0 + scol0);
        u32x4 a1 = *(const u32x4*)(Xn + (size_t)(rowA + srow1) * D_DIM + k0 + scol1);
        u32x4 b1 = *(const u32x4*)(Xn + (size_t)(rowB + srow1) * D_DIM + k0 + scol1);
        *(u32x4*)(&ldsA[0][srow0 * LDA_H + scol0]) = a0;
        *(u32x4*)(&ldsB[0][srow0 * LDA_H + scol0]) = b0;
        *(u32x4*)(&ldsA[0][srow1 * LDA_H + scol1]) = a1;
        *(u32x4*)(&ldsB[0][srow1 * LDA_H + scol1]) = b1;
        __syncthreads();
        mma_chunk(ldsA[0], ldsB[0], c, wr, wc, lm, lhi);
        __syncthreads();
    }
#endif

    // Epilogue: e = exp(s/T); zero diagonal; row sums (+ column sums for
    // off-diagonal tiles, by symmetry of S).
    const bool offdiag = (rt != ct);
#pragma unroll
    for (int m = 0; m < 4; ++m) {
#pragma unroll
        for (int n = 0; n < 2; ++n) {
            // C layout: VGPR r -> M = r + 8*lane_half ; N = lane%16
            const int gRowBase = rt * TILE + wr * 64 + m * 16 + lhi * 8;
            const int gcol     = ct * TILE + wc * 32 + n * 16 + lm;
            float colAcc = 0.0f;
#pragma unroll
            for (int r = 0; r < 8; ++r) {
                float e = __expf(c[m][n][r] * TEMP_INV);
                const int grow = gRowBase + r;
                if (grow == gcol) e = 0.0f;       // remove self-similarity
                colAcc += e;
                float s = e;                       // row sum over 16 lanes
                s += __shfl_xor(s, 8, 16);
                s += __shfl_xor(s, 4, 16);
                s += __shfl_xor(s, 2, 16);
                s += __shfl_xor(s, 1, 16);
                if (lm == 0) atomicAdd(&neg[grow], s);
            }
            if (offdiag) {
                colAcc += __shfl_xor(colAcc, 16, 32); // combine lane-halves
                if (lane < 16) atomicAdd(&neg[gcol], colAcc);
            }
        }
    }
}

// ---------------------------------------------------------------------------
// Kernel 3: loss = -log( mean_{2N}( pos / neg ) )
// ---------------------------------------------------------------------------
__global__ __launch_bounds__(256) void ntxent_final(
    const float* __restrict__ pos, const float* __restrict__ neg,
    float* __restrict__ out, int N)
{
    __shared__ float sh[256];
    const int t = threadIdx.x;
    float acc = 0.0f;
    for (int i = t; i < N; i += 256)
        acc += pos[i] * (1.0f / neg[i] + 1.0f / neg[i + N]);
    sh[t] = acc;
    __syncthreads();
    for (int off = 128; off > 0; off >>= 1) {
        if (t < off) sh[t] += sh[t + off];
        __syncthreads();
    }
    if (t == 0) out[0] = -__logf(sh[0] / (float)(2 * N));
}

// ---------------------------------------------------------------------------
extern "C" void kernel_launch(void* const* d_in, const int* in_sizes, int n_in,
                              void* d_out, int out_size, void* d_ws, size_t ws_size,
                              hipStream_t stream) {
    const float* y  = (const float*)d_in[0];
    const float* yh = (const float*)d_in[1];
    const int N  = in_sizes[0] / D_DIM;     // 4096
    const int N2 = 2 * N;                   // 8192

    _Float16* Xn = (_Float16*)d_ws;                                  // N2*512 halves (8 MB)
    float* neg = (float*)((char*)d_ws + (size_t)N2 * D_DIM * sizeof(_Float16));
    float* pos = neg + N2;

    ntxent_prep<<<dim3(N), dim3(256), 0, stream>>>(y, yh, Xn, pos, neg, N);

    const int Tt = N2 / TILE;                      // 64
    const int numWG = Tt * (Tt + 1) / 2;           // 2080 upper-tri tiles
    ntxent_simrows<<<dim3(numWG), dim3(256), 0, stream>>>(Xn, neg, Tt);

    ntxent_final<<<dim3(1), dim3(256), 0, stream>>>(pos, neg, (float*)d_out, N);
}
